// PathIntegrationModel_39539468927733
// MI455X (gfx1250) — compile-verified, matrix operand order checked by reading, not run on hardware
//
#include <hip/hip_runtime.h>
#include <hip/hip_bf16.h>

typedef __bf16 bf16;
typedef __attribute__((ext_vector_type(16))) __bf16 v16bf;
typedef __attribute__((ext_vector_type(8)))  float  v8f;
typedef __attribute__((ext_vector_type(4)))  unsigned int v4u;

#define H_   128
#define LIN_ 512
#define NPC_ 256
#define NHD_ 12
#define HS_  136   // lds_h row stride in bf16 (272B: +4 bank rotation/row)
#define FS_  520   // lds_feat row stride in bf16 (1040B: +4 bank rotation/row)
#define PS_  260   // lds_p row stride in f32

#define LOG2E_F 1.442695040888963f

union V16BF { v16bf v; v4u u[2]; };

// B-tile (or A-tile) from global f32, row-major [N,K]: 8 consecutive K values,
// then 8 more at +16 — matches CDNA5 16-bit WMMA operand striping.
__device__ __forceinline__ v16bf tile_from_global_f32(const float* __restrict__ p) {
  v16bf t;
#pragma unroll
  for (int e = 0; e < 8; ++e) t[e] = (bf16)p[e];
#pragma unroll
  for (int e = 0; e < 8; ++e) t[e + 8] = (bf16)p[16 + e];
  return t;
}

// A-tile from LDS bf16 (16B-aligned): two ds_load_b128.
__device__ __forceinline__ v16bf tile_from_lds(const bf16* p) {
  V16BF t;
  t.u[0] = *(const v4u*)(p);
  t.u[1] = *(const v4u*)(p + 16);
  return t.v;
}

// Branch-free fast exp (base-e) via v_exp_f32 (base-2 HW transcendental).
__device__ __forceinline__ float fast_exp(float x) {
  return __builtin_amdgcn_exp2f(x * LOG2E_F);
}

// Branch-free sigmoid: rcp(1 + exp2(-x*log2e)).
// x -> +inf: exp2->0, rcp(1)=1.  x -> -inf: exp2->inf, rcp(inf)=0.
__device__ __forceinline__ float fast_sigmoid(float x) {
  float e = __builtin_amdgcn_exp2f(x * -LOG2E_F);
  return __builtin_amdgcn_rcpf(1.0f + e);
}

// Hardware tanh if the builtin exists on this toolchain, else branch-free
// tanh(x) = 1 - 2*rcp(exp2(2x*log2e) + 1).
__device__ __forceinline__ float fast_tanh(float x) {
#if __has_builtin(__builtin_amdgcn_tanhf)
  return __builtin_amdgcn_tanhf(x);
#elif __has_builtin(__builtin_amdgcn_tanh_f32)
  return __builtin_amdgcn_tanh_f32(x);
#else
  float e = __builtin_amdgcn_exp2f(x * (2.0f * LOG2E_F));
  return 1.0f - 2.0f * __builtin_amdgcn_rcpf(e + 1.0f);
#endif
}

__global__ void __launch_bounds__(256, 1)
path_integration_lstm_kernel(
    const float* __restrict__ X,     const float* __restrict__ PC0,
    const float* __restrict__ HD0,
    const float* __restrict__ W_ih,  const float* __restrict__ W_hh,
    const float* __restrict__ b_ih,  const float* __restrict__ b_hh,
    const float* __restrict__ W_lin, const float* __restrict__ b_lin,
    const float* __restrict__ W_pc,  const float* __restrict__ b_pc,
    const float* __restrict__ W_hdo, const float* __restrict__ b_hdo,
    const float* __restrict__ w_cp,  const float* __restrict__ w_cd,
    const float* __restrict__ w_hp,  const float* __restrict__ w_hd,
    float* __restrict__ out_pc, float* __restrict__ out_hd,
    int T, int B)
{
  __shared__ __align__(16) bf16  lds_h[16 * HS_];     // h state, bf16, A-feed
  __shared__ __align__(16) bf16  lds_feat[16 * FS_];  // features, bf16
  __shared__ __align__(16) float lds_p[16 * PS_];     // pc0 staging / pc logits
  __shared__ __align__(16) float lds_d[16 * NHD_];    // hd0 staging / hd logits

  const int tid  = threadIdx.x;
  const int wave = tid >> 5;
  const int lane = tid & 31;
  const int nl   = lane & 15;   // N (or M for A-tile) within 16-tile
  const int hi   = lane >> 4;   // K-half selector
  const int rowbase = blockIdx.x * 16;
  const int hcol = wave * 16 + nl;   // this lane's H column (0..127)

  // ---------------- stage initial activations ----------------
  for (int idx = tid; idx < 16 * NPC_; idx += 256) {
    int m = idx >> 8, p = idx & 255;
    lds_p[m * PS_ + p] = PC0[(size_t)(rowbase + m) * NPC_ + p];
  }
  for (int idx = tid; idx < 16 * NHD_; idx += 256) {
    int m = idx / NHD_, p = idx % NHD_;
    lds_d[m * NHD_ + p] = HD0[(size_t)(rowbase + m) * NHD_ + p];
  }
  __syncthreads();

  // ---------------- h0, c0 (K=268, one-time, VALU) ----------------
  float cst[8], h0[8];
#pragma unroll
  for (int j = 0; j < 8; ++j) { cst[j] = 0.0f; h0[j] = 0.0f; }
  for (int p = 0; p < NPC_; ++p) {
    float wc = w_cp[(size_t)hcol * NPC_ + p];
    float wh = w_hp[(size_t)hcol * NPC_ + p];
#pragma unroll
    for (int j = 0; j < 8; ++j) {
      float a = lds_p[(j + 8 * hi) * PS_ + p];
      cst[j] = fmaf(a, wc, cst[j]);
      h0[j]  = fmaf(a, wh, h0[j]);
    }
  }
  for (int p = 0; p < NHD_; ++p) {
    float wc = w_cd[(size_t)hcol * NHD_ + p];
    float wh = w_hd[(size_t)hcol * NHD_ + p];
#pragma unroll
    for (int j = 0; j < 8; ++j) {
      float a = lds_d[(j + 8 * hi) * NHD_ + p];
      cst[j] = fmaf(a, wc, cst[j]);
      h0[j]  = fmaf(a, wh, h0[j]);
    }
  }
#pragma unroll
  for (int j = 0; j < 8; ++j)
    lds_h[(j + 8 * hi) * HS_ + hcol] = (bf16)h0[j];
  __syncthreads();

  // ---------------- per-lane loop invariants ----------------
  float wih[4][3], bias[4];
#pragma unroll
  for (int g = 0; g < 4; ++g) {
    int r = g * H_ + hcol;                    // W_hh / W_ih row (gate output)
    wih[g][0] = W_ih[r * 3 + 0];
    wih[g][1] = W_ih[r * 3 + 1];
    wih[g][2] = W_ih[r * 3 + 2];
    bias[g]   = b_ih[r] + b_hh[r];
  }

  // W_hh bf16 B-tiles pinned in registers: [gate][kblock], 128 VGPRs, T-fold reuse
  v16bf Bt[4][4];
#pragma unroll
  for (int g = 0; g < 4; ++g) {
    int r = g * H_ + hcol;
#pragma unroll
    for (int kb = 0; kb < 4; ++kb)
      Bt[g][kb] = tile_from_global_f32(W_hh + (size_t)r * H_ + kb * 32 + hi * 8);
  }

  // ---------------- LSTM recurrence ----------------
  for (int t = 0; t < T; ++t) {
    float xv[8][3];
#pragma unroll
    for (int j = 0; j < 8; ++j) {
      const float* xp = X + ((size_t)t * B + (rowbase + j + 8 * hi)) * 3;
      xv[j][0] = xp[0]; xv[j][1] = xp[1]; xv[j][2] = xp[2];
    }
    v8f acc[4];
#pragma unroll
    for (int g = 0; g < 4; ++g)
#pragma unroll
      for (int j = 0; j < 8; ++j)
        acc[g][j] = fmaf(xv[j][2], wih[g][2],
                    fmaf(xv[j][1], wih[g][1],
                    fmaf(xv[j][0], wih[g][0], bias[g])));

#pragma unroll
    for (int kb = 0; kb < 4; ++kb) {
      v16bf a = tile_from_lds(&lds_h[nl * HS_ + kb * 32 + hi * 8]);
#pragma unroll
      for (int g = 0; g < 4; ++g)
        acc[g] = __builtin_amdgcn_wmma_f32_16x16x32_bf16(
            false, a, false, Bt[g][kb], (short)0, acc[g], false, false);
    }

    float hn[8];
#pragma unroll
    for (int j = 0; j < 8; ++j) {
      float iv = fast_sigmoid(acc[0][j]);
      float fv = fast_sigmoid(acc[1][j]);
      float gv = fast_tanh(acc[2][j]);
      float ov = fast_sigmoid(acc[3][j]);
      float cn = fmaf(fv, cst[j], iv * gv);
      cst[j] = cn;
      hn[j]  = ov * fast_tanh(cn);
    }
    __syncthreads();                       // all waves done reading h_t
#pragma unroll
    for (int j = 0; j < 8; ++j)
      lds_h[(j + 8 * hi) * HS_ + hcol] = (bf16)hn[j];
    __syncthreads();                       // h_{t+1} visible
  }

  // ---------------- features = h @ W_lin.T + b_lin  (wave w -> cols [64w,64w+64)) ----
  v8f facc[4];
#pragma unroll
  for (int tt = 0; tt < 4; ++tt) {
    float bb = b_lin[(wave * 4 + tt) * 16 + nl];
#pragma unroll
    for (int j = 0; j < 8; ++j) facc[tt][j] = bb;
  }
#pragma unroll
  for (int kb = 0; kb < 4; ++kb) {
    v16bf a = tile_from_lds(&lds_h[nl * HS_ + kb * 32 + hi * 8]);
#pragma unroll
    for (int tt = 0; tt < 4; ++tt) {
      int r = (wave * 4 + tt) * 16 + nl;
      v16bf bt = tile_from_global_f32(W_lin + (size_t)r * H_ + kb * 32 + hi * 8);
      facc[tt] = __builtin_amdgcn_wmma_f32_16x16x32_bf16(
          false, a, false, bt, (short)0, facc[tt], false, false);
    }
  }
#pragma unroll
  for (int tt = 0; tt < 4; ++tt) {
    int n = (wave * 4 + tt) * 16 + nl;
#pragma unroll
    for (int j = 0; j < 8; ++j)
      lds_feat[(j + 8 * hi) * FS_ + n] = (bf16)facc[tt][j];
  }
  __syncthreads();

  // ---------------- pc logits = feat @ W_pc.T + b_pc (wave w -> cols [32w,32w+32)) ----
  v8f pacc[2];
#pragma unroll
  for (int tt = 0; tt < 2; ++tt) {
    float bb = b_pc[wave * 32 + tt * 16 + nl];
#pragma unroll
    for (int j = 0; j < 8; ++j) pacc[tt][j] = bb;
  }
#pragma unroll 4
  for (int kb = 0; kb < 16; ++kb) {
    v16bf a = tile_from_lds(&lds_feat[nl * FS_ + kb * 32 + hi * 8]);
#pragma unroll
    for (int tt = 0; tt < 2; ++tt) {
      int r = wave * 32 + tt * 16 + nl;
      v16bf bt = tile_from_global_f32(W_pc + (size_t)r * LIN_ + kb * 32 + hi * 8);
      pacc[tt] = __builtin_amdgcn_wmma_f32_16x16x32_bf16(
          false, a, false, bt, (short)0, pacc[tt], false, false);
    }
  }
#pragma unroll
  for (int tt = 0; tt < 2; ++tt) {
    int n = wave * 32 + tt * 16 + nl;
#pragma unroll
    for (int j = 0; j < 8; ++j)
      lds_p[(j + 8 * hi) * PS_ + n] = pacc[tt][j];
  }

  // ---------------- hd logits (N=12, VALU) ----------------
  if (tid < 16 * NHD_) {
    int m = tid / NHD_, n = tid % NHD_;
    float s = b_hdo[n];
    for (int k = 0; k < LIN_; ++k)
      s = fmaf((float)lds_feat[m * FS_ + k], W_hdo[(size_t)n * LIN_ + k], s);
    lds_d[m * NHD_ + n] = s;
  }
  __syncthreads();

  // ---------------- per-row softmax + store ----------------
  if (tid < 16) {
    int m = tid;
    size_t grow = (size_t)rowbase + m;
    // pc softmax (256); args <= 0 after max-subtract -> raw v_exp_f32 is safe
    float mx = -3.4e38f;
    for (int n = 0; n < NPC_; ++n) mx = fmaxf(mx, lds_p[m * PS_ + n]);
    float s = 0.0f;
    for (int n = 0; n < NPC_; ++n) {
      float e = fast_exp(lds_p[m * PS_ + n] - mx);
      lds_p[m * PS_ + n] = e;
      s += e;
    }
    float inv = 1.0f / s;
    for (int n = 0; n < NPC_; ++n)
      out_pc[grow * NPC_ + n] = lds_p[m * PS_ + n] * inv;
    // hd softmax (12)
    float mh = -3.4e38f;
    for (int n = 0; n < NHD_; ++n) mh = fmaxf(mh, lds_d[m * NHD_ + n]);
    float eh[NHD_], sh = 0.0f;
    for (int n = 0; n < NHD_; ++n) { eh[n] = fast_exp(lds_d[m * NHD_ + n] - mh); sh += eh[n]; }
    float invh = 1.0f / sh;
    for (int n = 0; n < NHD_; ++n)
      out_hd[grow * NHD_ + n] = eh[n] * invh;
  }
}

extern "C" void kernel_launch(void* const* d_in, const int* in_sizes, int n_in,
                              void* d_out, int out_size, void* d_ws, size_t ws_size,
                              hipStream_t stream) {
  (void)n_in; (void)out_size; (void)d_ws; (void)ws_size;
  const float* X     = (const float*)d_in[0];
  const float* PC0   = (const float*)d_in[1];
  const float* HD0   = (const float*)d_in[2];
  const float* W_ih  = (const float*)d_in[3];
  const float* W_hh  = (const float*)d_in[4];
  const float* b_ih  = (const float*)d_in[5];
  const float* b_hh  = (const float*)d_in[6];
  const float* W_lin = (const float*)d_in[7];
  const float* b_lin = (const float*)d_in[8];
  const float* W_pc  = (const float*)d_in[9];
  const float* b_pc  = (const float*)d_in[10];
  const float* W_hdo = (const float*)d_in[11];
  const float* b_hdo = (const float*)d_in[12];
  const float* w_cp  = (const float*)d_in[13];
  const float* w_cd  = (const float*)d_in[14];
  const float* w_hp  = (const float*)d_in[15];
  const float* w_hd  = (const float*)d_in[16];

  const int B = in_sizes[1] / NPC_;         // initial_pc_activation: [B, 256]
  const int T = in_sizes[0] / (B * 3);      // velocity_inputs: [T, B, 3]

  float* out_pc = (float*)d_out;
  float* out_hd = out_pc + (size_t)B * NPC_;

  dim3 grid(B / 16), block(256);
  hipLaunchKernelGGL(path_integration_lstm_kernel, grid, block, 0, stream,
                     X, PC0, HD0, W_ih, W_hh, b_ih, b_hh, W_lin, b_lin,
                     W_pc, b_pc, W_hdo, b_hdo, w_cp, w_cd, w_hp, w_hd,
                     out_pc, out_hd, T, B);
}
